// GAT_29712583754275
// MI455X (gfx1250) — matrix operand order repaired and checked by reading, not compiled, not run on hardware
//
#include <hip/hip_runtime.h>
#include <hip/hip_bf16.h>

// ---------------------------------------------------------------------------
// GAT 3-layer + 2 MLP heads for MI455X (gfx1250, wave32, WMMA).
// GEMMs use v_wmma_f32_16x16x32_bf16 (bf16 A/B, f32 accumulate).
// Edge softmax/aggregation uses coalesced gather + f32 atomics (HBM-bound).
// ---------------------------------------------------------------------------

#define GAT_IN    512
#define GAT_C     64
#define NEG_SLOPE 0.2f

typedef __attribute__((ext_vector_type(16))) __bf16          v16bf;
typedef __attribute__((ext_vector_type(8)))  float           v8f;
typedef __attribute__((ext_vector_type(8)))  unsigned short  u16x8;

// ---------------------------------- helpers --------------------------------

__device__ __forceinline__ unsigned short f32_to_bf16_bits(float x) {
    unsigned u = __float_as_uint(x);
    unsigned r = u + 0x7FFFu + ((u >> 16) & 1u);   // round-to-nearest-even
    return (unsigned short)(r >> 16);
}

__device__ __forceinline__ float leaky(float x) {
    return x > 0.f ? x : NEG_SLOPE * x;
}

// float atomic max via int/uint monotonic-bit trick (safe for mixed signs)
__device__ __forceinline__ void atomicMaxF(float* addr, float val) {
    if (val >= 0.f) atomicMax((int*)addr, __float_as_int(val));
    else            atomicMin((unsigned int*)addr, __float_as_uint(val));
}

// host-side -inf constant (can't use device __uint_as_float here)
static inline float host_neg_inf() {
    union { unsigned u; float f; } c; c.u = 0xFF800000u; return c.f;
}

// ------------------------------- small kernels -----------------------------

__global__ void cvt_bf16_kernel(const float* __restrict__ in,
                                unsigned short* __restrict__ out, int count) {
    int i = blockIdx.x * blockDim.x + threadIdx.x;
    if (i < count) out[i] = f32_to_bf16_bits(in[i]);
}

__global__ void fill_f32_kernel(float* __restrict__ p, float v, int count) {
    int i = blockIdx.x * blockDim.x + threadIdx.x;
    if (i < count) p[i] = v;
}

// Pack row-major W[K,Ncols] (f32) into WMMA B-fragment order (bf16):
// Wpk[(((kb*NT)+nb)*32 + lane)*16 + e]  where lane 0-15 hold K {0..7,16..23} of the
// 32-deep K-slab, lanes 16-31 hold K {8..15,24..31} (ISA 16-bit layout).
__global__ void pack_w_kernel(const float* __restrict__ W,
                              unsigned short* __restrict__ Wpk,
                              int Kd, int Ncols) {
    int NT = Ncols >> 4;
    int total = (Kd >> 5) * NT * 512;          // 32 lanes * 16 elems per tile
    int idx = blockIdx.x * blockDim.x + threadIdx.x;
    if (idx >= total) return;
    int e    = idx & 15;
    int lane = (idx >> 4) & 31;
    int tile = idx >> 9;
    int nb   = tile % NT;
    int kb   = tile / NT;
    int k = kb * 32 + ((lane >> 4) << 3) + (e & 7) + ((e >> 3) << 4);
    int n = nb * 16 + (lane & 15);
    Wpk[idx] = f32_to_bf16_bits(W[(size_t)k * Ncols + n]);
}

// ------------------------------- WMMA GEMM ---------------------------------
// C[M, NT*16] = A[M, Kd] (bf16 bits, row-major) x Bpk (pre-packed fragments).
// One wave per 16-row M tile; NT accumulators (NT = 8 or 4), K-loop of 32.
// Optional fused bias + ReLU epilogue for the MLP heads.

template <int NT>
__global__ __launch_bounds__(128)
void wmma_gemm_kernel(const unsigned short* __restrict__ A,
                      const unsigned short* __restrict__ Bpk,
                      float* __restrict__ C,
                      int M, int Kd,
                      const float* __restrict__ bias, int do_relu) {
    const int lane = threadIdx.x & 31;
    const int wave = threadIdx.x >> 5;
    const int mt   = blockIdx.x * 4 + wave;
    if (mt * 16 >= M) return;

    const int m0  = mt * 16;
    const int hi  = lane >> 4;                 // half-select within the wave
    const int row = m0 + (lane & 15);          // A row owned by this lane
    const int KT  = Kd >> 5;

    v8f acc[NT];
#pragma unroll
    for (int nb = 0; nb < NT; ++nb)
        acc[nb] = (v8f){0.f, 0.f, 0.f, 0.f, 0.f, 0.f, 0.f, 0.f};

    // ISA 16-bit A layout: lanes<16 hold K {0..7, 16..23}; lanes>=16 offset +8
    const unsigned short* arow = A + (size_t)row * Kd + hi * 8;

    for (int kb = 0; kb < KT; ++kb) {
        union { u16x8 h[2]; v16bf v; } af;
        af.h[0] = *(const u16x8*)(arow + kb * 32);        // 16B load
        af.h[1] = *(const u16x8*)(arow + kb * 32 + 16);   // 16B load
        const unsigned short* bp = Bpk + ((size_t)kb * NT * 32 + lane) * 16;
#pragma unroll
        for (int nb = 0; nb < NT; ++nb) {
            v16bf bf = *(const v16bf*)(bp + (size_t)nb * 512);   // 32B load
            acc[nb] = __builtin_amdgcn_wmma_f32_16x16x32_bf16(
                false, af.v, false, bf, (short)0, acc[nb], false, false);
        }
    }

    // C/D layout: VGPR r -> M = r + 8*hi ; N = lane & 15
    const int Ncols  = NT * 16;
    const int n_lane = lane & 15;
#pragma unroll
    for (int nb = 0; nb < NT; ++nb) {
        int n = nb * 16 + n_lane;
        float bval = bias ? bias[n] : 0.f;
#pragma unroll
        for (int r = 0; r < 8; ++r) {
            int m = m0 + hi * 8 + r;
            float v = acc[nb][r] + bval;
            if (do_relu) v = fmaxf(v, 0.f);
            C[(size_t)m * Ncols + n] = v;
        }
    }
}

// ------------------------------ GAT edge phase -----------------------------

// s[n,h] = <h[n,h,:], a_src[h,:]> ; t likewise. One thread per (node, head).
__global__ void score_kernel(const float* __restrict__ h,
                             const float* __restrict__ a_src,
                             const float* __restrict__ a_dst,
                             float* __restrict__ s, float* __restrict__ t,
                             int n, int H) {
    int i = blockIdx.x * blockDim.x + threadIdx.x;
    if (i >= n * H) return;
    int head = i % H, node = i / H;
    const float* hp = h + ((size_t)node * H + head) * GAT_C;
    const float* as = a_src + head * GAT_C;
    const float* ad = a_dst + head * GAT_C;
    float ss = 0.f, tt = 0.f;
#pragma unroll 8
    for (int c = 0; c < GAT_C; ++c) { float v = hp[c]; ss += v * as[c]; tt += v * ad[c]; }
    s[i] = ss;
    t[i] = tt;
}

// Edges e < E come from edge_index; e >= E are the appended self-loops.
__global__ void edge_max_kernel(const int* __restrict__ src, const int* __restrict__ dst,
                                int E, int n,
                                const float* __restrict__ s, const float* __restrict__ t,
                                float* __restrict__ m, int H) {
    int e = blockIdx.x * blockDim.x + threadIdx.x;
    if (e >= E + n) return;
    int se, de;
    if (e < E) { se = src[e]; de = dst[e]; } else { se = de = e - E; }
    for (int h = 0; h < H; ++h) {
        float v = leaky(s[se * H + h] + t[de * H + h]);
        atomicMaxF(&m[de * H + h], v);
    }
}

__global__ void edge_expsum_kernel(const int* __restrict__ src, const int* __restrict__ dst,
                                   int E, int n,
                                   const float* __restrict__ s, const float* __restrict__ t,
                                   const float* __restrict__ m,
                                   float* __restrict__ den, float* __restrict__ ex,
                                   int H) {
    int e = blockIdx.x * blockDim.x + threadIdx.x;
    if (e >= E + n) return;
    int se, de;
    if (e < E) { se = src[e]; de = dst[e]; } else { se = de = e - E; }
    for (int h = 0; h < H; ++h) {
        float v  = leaky(s[se * H + h] + t[de * H + h]);
        float xv = __expf(v - m[de * H + h]);
        ex[e * H + h] = xv;
        atomicAdd(&den[de * H + h], xv);
    }
}

// One thread per (edge, feature): coalesced gather of h[src] rows and
// coalesced atomic scatter into agg[dst] rows. HC = H*64.
__global__ void edge_agg_kernel(const int* __restrict__ src, const int* __restrict__ dst,
                                int E, int n,
                                const float* __restrict__ h,
                                const float* __restrict__ ex, const float* __restrict__ den,
                                float* __restrict__ agg, int H, int HC) {
    int idx = blockIdx.x * blockDim.x + threadIdx.x;
    int total = (E + n) * HC;
    if (idx >= total) return;
    int e = idx / HC;
    int f = idx - e * HC;
    int head = f >> 6;                     // C == 64
    int se, de;
    if (e < E) { se = src[e]; de = dst[e]; } else { se = de = e - E; }
    float alpha = ex[e * H + head] / (den[de * H + head] + 1e-16f);
    atomicAdd(&agg[(size_t)de * HC + f], h[(size_t)se * HC + f] * alpha);
}

__global__ void finalize_kernel(const float* __restrict__ agg,
                                const float* __restrict__ b,
                                float* __restrict__ out,
                                int n, int HC, int do_relu) {
    int i = blockIdx.x * blockDim.x + threadIdx.x;
    if (i >= n * HC) return;
    float v = agg[i] + b[i % HC];
    if (do_relu) v = fmaxf(v, 0.f);
    out[i] = v;
}

// ------------------------------- host driver -------------------------------

static inline int cdiv(int a, int b) { return (a + b - 1) / b; }

static void launch_gemm(const unsigned short* A, const unsigned short* Bpk, float* C,
                        int M, int Kd, int NT, const float* bias, int do_relu,
                        hipStream_t stream) {
    int blocks = cdiv(cdiv(M, 16), 4);
    if (NT == 8)      wmma_gemm_kernel<8><<<blocks, 128, 0, stream>>>(A, Bpk, C, M, Kd, bias, do_relu);
    else /* NT==4 */  wmma_gemm_kernel<4><<<blocks, 128, 0, stream>>>(A, Bpk, C, M, Kd, bias, do_relu);
}

// One full GAT layer: x_in[n,F_in] -> out[n, H*64]
static void run_gat_layer(const float* x_in, int F_in, int H,
                          const float* W, const float* a_src, const float* a_dst,
                          const float* b,
                          const int* src, const int* dst, int E, int n,
                          float* h_lin, float* agg, float* out, int do_relu,
                          unsigned short* Abf, unsigned short* Wpk,
                          float* sbuf, float* tbuf, float* mbuf, float* dbuf,
                          float* exbuf, hipStream_t stream) {
    const int HC = H * GAT_C;
    const int NT = HC / 16;
    const int Etot = E + n;

    cvt_bf16_kernel<<<cdiv(n * F_in, 256), 256, 0, stream>>>(x_in, Abf, n * F_in);
    {
        int total = (F_in / 32) * NT * 512;
        pack_w_kernel<<<cdiv(total, 256), 256, 0, stream>>>(W, Wpk, F_in, HC);
    }
    launch_gemm(Abf, Wpk, h_lin, n, F_in, NT, nullptr, 0, stream);

    score_kernel<<<cdiv(n * H, 256), 256, 0, stream>>>(h_lin, a_src, a_dst, sbuf, tbuf, n, H);

    fill_f32_kernel<<<cdiv(n * H, 256), 256, 0, stream>>>(mbuf, host_neg_inf(), n * H);
    fill_f32_kernel<<<cdiv(n * H, 256), 256, 0, stream>>>(dbuf, 0.f, n * H);
    fill_f32_kernel<<<cdiv(n * HC, 256), 256, 0, stream>>>(agg, 0.f, n * HC);

    edge_max_kernel<<<cdiv(Etot, 256), 256, 0, stream>>>(src, dst, E, n, sbuf, tbuf, mbuf, H);
    edge_expsum_kernel<<<cdiv(Etot, 256), 256, 0, stream>>>(src, dst, E, n, sbuf, tbuf, mbuf, dbuf, exbuf, H);
    edge_agg_kernel<<<cdiv(Etot * HC, 256), 256, 0, stream>>>(src, dst, E, n, h_lin, exbuf, dbuf, agg, H, HC);

    finalize_kernel<<<cdiv(n * HC, 256), 256, 0, stream>>>(agg, b, out, n, HC, do_relu);
}

extern "C" void kernel_launch(void* const* d_in, const int* in_sizes, int n_in,
                              void* d_out, int out_size, void* d_ws, size_t ws_size,
                              hipStream_t stream) {
    // ---- inputs (setup_inputs order) ----
    const float* x       = (const float*)d_in[0];
    const int*   eidx    = (const int*)  d_in[1];
    const float* W0      = (const float*)d_in[2];
    const float* a_src0  = (const float*)d_in[3];
    const float* a_dst0  = (const float*)d_in[4];
    const float* b0      = (const float*)d_in[5];
    const float* W1      = (const float*)d_in[6];
    const float* a_src1  = (const float*)d_in[7];
    const float* a_dst1  = (const float*)d_in[8];
    const float* b1      = (const float*)d_in[9];
    const float* W2      = (const float*)d_in[10];
    const float* a_src2  = (const float*)d_in[11];
    const float* a_dst2  = (const float*)d_in[12];
    const float* b2      = (const float*)d_in[13];
    const float* Wv      = (const float*)d_in[14];
    const float* bv      = (const float*)d_in[15];
    const float* Wt      = (const float*)d_in[16];
    const float* bt      = (const float*)d_in[17];

    const int N = in_sizes[0] / GAT_IN;      // 50000
    const int E = in_sizes[1] / 2;           // 800000
    const int* src = eidx;                   // edge_index[0]
    const int* dst = eidx + E;               // edge_index[1]

    // ---- workspace carve-out (~111 MB) ----
    char*  ws  = (char*)d_ws;
    size_t off = 0;
    auto carve = [&](size_t bytes) -> void* {
        void* p = (void*)(ws + off);
        off += (bytes + 255) & ~(size_t)255;
        return p;
    };
    unsigned short* Abf   = (unsigned short*)carve((size_t)N * GAT_IN * 2);   // bf16 activations
    unsigned short* Wpk   = (unsigned short*)carve(512 * 1024);               // packed weights
    float*          hA    = (float*)carve((size_t)N * 128 * 4);               // linear out / layer h
    float*          hB    = (float*)carve((size_t)N * 128 * 4);               // aggregation accum
    float*          sbuf  = (float*)carve((size_t)N * 2 * 4);
    float*          tbuf  = (float*)carve((size_t)N * 2 * 4);
    float*          mbuf  = (float*)carve((size_t)N * 2 * 4);
    float*          dbuf  = (float*)carve((size_t)N * 2 * 4);
    float*          exbuf = (float*)carve((size_t)(E + N) * 2 * 4);
    (void)ws_size; (void)n_in; (void)out_size;

    float* out_h = (float*)d_out;                  // [N,64]
    float* out_v = out_h + (size_t)N * GAT_C;      // [N,64]
    float* out_t = out_v + (size_t)N * GAT_C;      // [N,64]

    // ---- layer 0: 512 -> 2x64, relu; agg into hB, result into hA ----
    run_gat_layer(x, GAT_IN, 2, W0, a_src0, a_dst0, b0, src, dst, E, N,
                  hA, hB, hA, /*relu=*/1, Abf, Wpk, sbuf, tbuf, mbuf, dbuf, exbuf, stream);

    // ---- layer 1: 128 -> 2x64, relu ----
    run_gat_layer(hA, 128, 2, W1, a_src1, a_dst1, b1, src, dst, E, N,
                  hA, hB, hA, /*relu=*/1, Abf, Wpk, sbuf, tbuf, mbuf, dbuf, exbuf, stream);

    // ---- layer 2: 128 -> 1x64, no relu; write h straight to d_out ----
    run_gat_layer(hA, 128, 1, W2, a_src2, a_dst2, b2, src, dst, E, N,
                  hA, hB, out_h, /*relu=*/0, Abf, Wpk, sbuf, tbuf, mbuf, dbuf, exbuf, stream);

    // ---- MLP heads: relu(h @ Wv + bv), relu(h @ Wt + bt), fused epilogue ----
    cvt_bf16_kernel<<<cdiv(N * GAT_C, 256), 256, 0, stream>>>(out_h, Abf, N * GAT_C);
    {
        int total = (GAT_C / 32) * 4 * 512;  // K=64 -> KT=2, NT=4
        pack_w_kernel<<<cdiv(total, 256), 256, 0, stream>>>(Wv, Wpk, GAT_C, GAT_C);
        launch_gemm(Abf, Wpk, out_v, N, GAT_C, 4, bv, /*relu=*/1, stream);
        pack_w_kernel<<<cdiv(total, 256), 256, 0, stream>>>(Wt, Wpk, GAT_C, GAT_C);
        launch_gemm(Abf, Wpk, out_t, N, GAT_C, 4, bt, /*relu=*/1, stream);
    }
}
